// ionCNN_25615184953367
// MI455X (gfx1250) — matrix operand order repaired and checked by reading, not compiled, not run on hardware
//
#include <hip/hip_runtime.h>
#include <cstdint>
#include <cstddef>

// ---------------- problem sizes ----------------
static constexpr int B_   = 16;
static constexpr int S_   = 2048;
static constexpr int VI_  = 208;     // V*I
static constexpr int VIP_ = 224;     // VI padded to multiple of 32 (K for L1)
static constexpr int P_   = B_ * S_; // 32768 positions
static constexpr int O1_  = 512;     // NU
static constexpr int O2_  = 1024;    // 2NU
static constexpr int O3_  = 2048;    // 4NU
static constexpr int FJ1_ = 1024;    // fc1 out (2NU)
static constexpr int FK1_ = 2048;    // fc1 in  (4NU)
static constexpr int FJ2_ = 512;     // fc2 out (NU)
static constexpr int FK2_ = 1024;    // fc2 in  (2NU)
static constexpr int NSEG_ = 64;     // stats partial segments (generic path)
static constexpr float EPSF = 1e-5f;

// GEMM tiling
static constexpr int WTILE_E = 256 * 32;  // weight LDS tile elems (16KB)
static constexpr int ATILE_E = 64 * 32;   // activation LDS tile elems (4KB)
static constexpr unsigned LDS_A_BASE = (unsigned)(2 * WTILE_E) * 2u;  // byte offset of A tiles

// ---------------- WMMA / TDM vector types ----------------
typedef __bf16 v16bf __attribute__((ext_vector_type(16)));
typedef __bf16 v8bf  __attribute__((ext_vector_type(8)));
typedef float  v8f   __attribute__((ext_vector_type(8)));
typedef unsigned int v4u __attribute__((ext_vector_type(4)));
typedef int v8i __attribute__((ext_vector_type(8)));
typedef int v4i __attribute__((ext_vector_type(4)));

// ---------------- helpers ----------------
__device__ __forceinline__ unsigned short f2bf(float f) {
  unsigned u = __float_as_uint(f);
  u += 0x7FFFu + ((u >> 16) & 1u);   // RNE
  return (unsigned short)(u >> 16);
}
// monotone float<->uint order mapping (for deterministic atomic min/max)
__device__ __forceinline__ unsigned f2ord(float f) {
  unsigned u = __float_as_uint(f);
  return (u & 0x80000000u) ? ~u : (u | 0x80000000u);
}
__device__ __forceinline__ float ord2f(unsigned e) {
  unsigned u = (e & 0x80000000u) ? (e & 0x7FFFFFFFu) : ~e;
  return __uint_as_float(u);
}

// Tensor Data Mover: 2D tile (tile_d0 x tile_d1, 2-byte elems) global -> LDS.
// D# bitfields per CDNA5 ISA (count=1, type=2).
// 6-arg builtin form (this toolchain): (v4u, v8i, v4i, v4i, v8i, i32 cpol).
__device__ __forceinline__ void tdm_load_2d(unsigned lds_byte_off, const void* gptr,
                                            unsigned tensor_d0, unsigned tensor_d1,
                                            unsigned tile_d0, unsigned tile_d1,
                                            unsigned stride_elems) {
  unsigned long long ga = (unsigned long long)gptr;
  v4u g0;
  g0[0] = 1u;                                              // count=1, user mode
  g0[1] = lds_byte_off;                                    // lds_addr
  g0[2] = (unsigned)(ga & 0xFFFFFFFFu);                    // global_addr[31:0]
  g0[3] = (unsigned)((ga >> 32) & 0x01FFFFFFu) | (2u << 30); // addr[56:32] | type=2
  v8i g1;
  g1[0] = (int)(1u << 16);                                 // data_size = 1 -> 2 bytes
  g1[1] = (int)((tensor_d0 & 0xFFFFu) << 16);              // tensor_dim0[15:0] @ bit48
  g1[2] = (int)(((tensor_d1 & 0xFFFFu) << 16) | ((tensor_d0 >> 16) & 0xFFFFu));
  g1[3] = (int)(((tile_d0 & 0xFFFFu) << 16) | ((tensor_d1 >> 16) & 0xFFFFu));
  g1[4] = (int)(tile_d1 & 0xFFFFu);                        // tile_dim1, tile_dim2=0
  g1[5] = (int)stride_elems;                               // tensor_dim0_stride[31:0]
  g1[6] = 0;
  g1[7] = 0;
  v4i g2 = {0, 0, 0, 0};
  v4i g3 = {0, 0, 0, 0};
  v8i g4 = {0, 0, 0, 0, 0, 0, 0, 0};
  __builtin_amdgcn_tensor_load_to_lds(g0, g1, g2, g3, g4, 0);
}

// ---------------- feature kernel: feat[p][c] ----------------
__global__ void feat_kernel(const float* __restrict__ obs,
                            const float* __restrict__ th,
                            const float* __restrict__ dsf,
                            float* __restrict__ feat) {
  int idx = blockIdx.x * blockDim.x + threadIdx.x;
  if (idx >= P_ * VI_) return;
  int c = idx % VI_;
  int p = idx / VI_;
  int b = p >> 11;               // p / S_
  float o  = obs[p];
  float t0 = th[b * VI_ + c];
  float scale = (1.0f / (1.0f + __expf(-dsf[0]))) * 0.2f;
  float best = 0.0f;
#pragma unroll
  for (int si = 0; si < 3; ++si) {
    float sh = (si == 0) ? 0.0f : ((si == 1) ? -1.0f : 1.0f);
    float t = t0 + sh;
    float ppm = (o - t) / (t + 1e-6f) * 1e6f;
    float l = __expf(-fabsf(ppm * scale));
    best = fmaxf(best, l);
  }
  feat[idx] = best;
}

// ---------------- weight f32 -> bf16 (with K zero-pad) ----------------
__global__ void wcvt_kernel(const float* __restrict__ Wf,
                            unsigned short* __restrict__ Wb,
                            int O, int K, int Kpad) {
  int idx = blockIdx.x * blockDim.x + threadIdx.x;
  if (idx >= O * Kpad) return;
  int k = idx % Kpad, o = idx / Kpad;
  float v = (k < K) ? Wf[(size_t)o * K + k] : 0.0f;
  Wb[idx] = f2bf(v);
}

// ---------------- BN stats: deterministic segmented reduction ----------------
__global__ void stats_partial_kernel(const float* __restrict__ Y,
                                     float* __restrict__ psum, float* __restrict__ psq,
                                     int O, int chunk) {
  int o = blockIdx.x * blockDim.x + threadIdx.x;
  if (o >= O) return;
  int p0 = blockIdx.y * chunk;
  float s = 0.0f, q = 0.0f;
  for (int p = p0; p < p0 + chunk; ++p) {
    float v = Y[(size_t)p * O + o];
    s += v; q += v * v;
  }
  psum[(size_t)blockIdx.y * O + o] = s;
  psq [(size_t)blockIdx.y * O + o] = q;
}

__global__ void stats_final_kernel(const float* __restrict__ psum,
                                   const float* __restrict__ psq,
                                   int nseg, int O, float invP,
                                   float* __restrict__ mean, float* __restrict__ rstd) {
  int o = blockIdx.x * blockDim.x + threadIdx.x;
  if (o >= O) return;
  float s = 0.0f, q = 0.0f;
  for (int i = 0; i < nseg; ++i) {
    s += psum[(size_t)i * O + o];
    q += psq [(size_t)i * O + o];
  }
  float m = s * invP;
  float v = q * invP - m * m;
  mean[o] = m;
  rstd[o] = rsqrtf(v + EPSF);
}

// ---------------- BN apply (+optional relu), f32 -> bf16 with pad ----------------
__global__ void bnact_kernel(const float* __restrict__ Y,
                             const float* __restrict__ mean, const float* __restrict__ rstd,
                             const float* __restrict__ g, const float* __restrict__ be,
                             unsigned short* __restrict__ X,
                             int O, int Opad, int do_relu) {
  size_t idx = (size_t)blockIdx.x * blockDim.x + threadIdx.x;
  if (idx >= (size_t)P_ * Opad) return;
  int o = (int)(idx % Opad);
  size_t p = idx / Opad;
  float v = 0.0f;
  if (o < O) {
    v = g[o] * (Y[p * O + o] - mean[o]) * rstd[o] + be[o];
    if (do_relu) v = fmaxf(v, 0.0f);
  }
  X[idx] = f2bf(v);
}

// ---------------- bf16 WMMA GEMM: Y[p][o] = sum_k X[p][k] * W[o][k] ----------------
// Data path: TDM (tensor_load_to_lds) double-buffers a 256x32 weight tile and a
// 64x32 activation tile in LDS; waves read WMMA fragments from LDS (ds_load_b128).
// MODE 0: store f32 Y.  MODE 1: fused reduce epilogue (per-channel sum/sumsq segments
//         + per-(batch,channel) ordered-atomic min/max of preactivation) — no Y store.
template <int MODE>
__global__ __launch_bounds__(256) void gemm_bf16_kernel(
    const unsigned short* __restrict__ X, const unsigned short* __restrict__ W,
    float* __restrict__ Y,
    float* __restrict__ psum, float* __restrict__ psq,
    unsigned* __restrict__ umax, unsigned* __restrict__ umin,
    int K, int O) {
  // Single shared object -> LDS offset 0; TDM byte offsets are compile-time constants.
  __shared__ unsigned short smem[2 * WTILE_E + 2 * ATILE_E];  // 40KB

  const int tid  = threadIdx.x;
  const int wv   = tid >> 5;      // wave 0..7
  const int lane = tid & 31;
  const int n    = lane & 15;     // row (A) / col (B,D) within 16
  const int h    = lane >> 4;     // half selector

  const int pBlock = blockIdx.x * 64;
  const int oBlock = blockIdx.y * 256;
  const int pWave  = pBlock + (wv & 1) * 32;   // 32 p-rows per wave
  const int oWave  = oBlock + (wv >> 1) * 64;  // 64 o-cols per wave

  v8f acc[2][4];
#pragma unroll
  for (int mt = 0; mt < 2; ++mt)
#pragma unroll
    for (int nt = 0; nt < 4; ++nt)
#pragma unroll
      for (int r = 0; r < 8; ++r) acc[mt][nt][r] = 0.0f;

  const int nsteps = K / 32;

  if (wv == 0) {   // prologue: buffer 0 (W tile + A tile)
    tdm_load_2d(0u, W + (size_t)oBlock * K, K, O, 32, 256, K);
    tdm_load_2d(LDS_A_BASE, X + (size_t)pBlock * K, K, P_, 32, 64, K);
  }

  for (int i = 0; i < nsteps; ++i) {
    const int buf = i & 1;
    if (wv == 0) {
      if (i + 1 < nsteps) {
        const int nbuf = buf ^ 1;
        const int k1 = (i + 1) * 32;
        tdm_load_2d((unsigned)(nbuf * WTILE_E * 2),
                    W + (size_t)oBlock * K + k1, K, O, 32, 256, K);
        tdm_load_2d(LDS_A_BASE + (unsigned)(nbuf * ATILE_E * 2),
                    X + (size_t)pBlock * K + k1, K, P_, 32, 64, K);
        __builtin_amdgcn_s_wait_tensorcnt((short)2);  // pair i complete; pair i+1 in flight
      } else {
        __builtin_amdgcn_s_wait_tensorcnt((short)0);
      }
    }
    __syncthreads();   // LDS tiles for step i visible to all waves

    const unsigned short* wBase = &smem[buf * WTILE_E];
    const unsigned short* aBase = &smem[2 * WTILE_E + buf * ATILE_E];

    // A fragments (16x32): lanes 0-15 K 0..15, lanes 16-31 K 16..31
    v16bf a[2];
#pragma unroll
    for (int mt = 0; mt < 2; ++mt) {
      const unsigned short* xr = aBase + ((wv & 1) * 32 + mt * 16 + n) * 32 + h * 16;
      v8bf lo = *reinterpret_cast<const v8bf*>(xr);
      v8bf hi = *reinterpret_cast<const v8bf*>(xr + 8);
#pragma unroll
      for (int j = 0; j < 8; ++j) { a[mt][j] = lo[j]; a[mt][8 + j] = hi[j]; }
    }
    // B fragments (32x16): lane n = column, 16 contiguous K per half
    v16bf bfr[4];
#pragma unroll
    for (int nt = 0; nt < 4; ++nt) {
      const unsigned short* wr = wBase + ((wv >> 1) * 64 + nt * 16 + n) * 32 + h * 16;
      bfr[nt] = *reinterpret_cast<const v16bf*>(wr);
    }
#pragma unroll
    for (int mt = 0; mt < 2; ++mt)
#pragma unroll
      for (int nt = 0; nt < 4; ++nt)
        acc[mt][nt] = __builtin_amdgcn_wmma_f32_16x16x32_bf16(
            false, a[mt], false, bfr[nt], (short)0, acc[mt][nt], false, false);

    __syncthreads();   // all waves done with buf before it is overwritten at step i+2
  }

  if constexpr (MODE == 0) {
    const size_t strideO = (size_t)O;
    float* yp = Y + (size_t)(pWave + h * 8) * strideO + (oWave + n);
#pragma unroll
    for (int mt = 0; mt < 2; ++mt) {
#pragma unroll
      for (int r = 0; r < 8; ++r) {
        float* rowp = yp + (size_t)(mt * 16 + r) * strideO;
#pragma unroll
        for (int nt = 0; nt < 4; ++nt) rowp[nt * 16] = acc[mt][nt][r];
      }
    }
  } else {
    const int seg = blockIdx.x * 2 + (wv & 1);   // 32-row p-segment id (deterministic)
    const int b   = pWave >> 11;                 // whole block within one batch row
#pragma unroll
    for (int nt = 0; nt < 4; ++nt) {
      float s = 0.0f, q = 0.0f, mx = -3.4e38f, mn = 3.4e38f;
#pragma unroll
      for (int mt = 0; mt < 2; ++mt)
#pragma unroll
        for (int r = 0; r < 8; ++r) {
          float v = acc[mt][nt][r];
          s += v; q += v * v;
          mx = fmaxf(mx, v); mn = fminf(mn, v);
        }
      // combine the two lane-halves holding the same output column
      s  += __shfl_xor(s, 16, 32);
      q  += __shfl_xor(q, 16, 32);
      mx  = fmaxf(mx, __shfl_xor(mx, 16, 32));
      mn  = fminf(mn, __shfl_xor(mn, 16, 32));
      if (lane < 16) {
        int o = oWave + nt * 16 + n;
        psum[(size_t)seg * O + o] = s;
        psq [(size_t)seg * O + o] = q;
        atomicMax(&umax[b * O + o], f2ord(mx));
        atomicMin(&umin[b * O + o], f2ord(mn));
      }
    }
  }
}

// ---------------- L3 minmax init ----------------
__global__ void init_minmax_kernel(unsigned* __restrict__ umax, unsigned* __restrict__ umin, int nelem) {
  int i = blockIdx.x * blockDim.x + threadIdx.x;
  if (i < nelem) { umax[i] = 0u; umin[i] = 0xFFFFFFFFu; }
}

// ---------------- L3: max_s relu(BN(y)) from min/max of preactivation ----------------
__global__ void bnmax_kernel(const unsigned* __restrict__ umax, const unsigned* __restrict__ umin,
                             const float* __restrict__ mean, const float* __restrict__ rstd,
                             const float* __restrict__ g, const float* __restrict__ be,
                             float* __restrict__ M, int O) {
  int idx = blockIdx.x * blockDim.x + threadIdx.x;   // b*O + o
  if (idx >= B_ * O) return;
  int o = idx % O;
  float slope = g[o] * rstd[o];
  float y = (slope >= 0.0f) ? ord2f(umax[idx]) : ord2f(umin[idx]);
  float v = slope * (y - mean[o]) + be[o];
  M[idx] = fmaxf(v, 0.0f);
}

// ---------------- small FC: Z[b][j] = sum_k A[b][k] * W[j][k] ----------------
__global__ void fc_kernel(const float* __restrict__ A, const float* __restrict__ W,
                          float* __restrict__ Z, int Kdim, int J) {
  int idx = blockIdx.x * blockDim.x + threadIdx.x;   // b*J + j
  if (idx >= B_ * J) return;
  int b = idx / J, j = idx % J;
  const float* ar = A + (size_t)b * Kdim;
  const float* wr = W + (size_t)j * Kdim;
  float s = 0.0f;
  for (int k = 0; k < Kdim; ++k) s += ar[k] * wr[k];
  Z[idx] = s;
}

// ---------------- BN over batch axis (16 values) + relu ----------------
__global__ void bnrelu_c_kernel(const float* __restrict__ Z,
                                const float* __restrict__ g, const float* __restrict__ be,
                                float* __restrict__ Aout, int J) {
  int j = blockIdx.x * blockDim.x + threadIdx.x;
  if (j >= J) return;
  float s = 0.0f, q = 0.0f;
  for (int b = 0; b < B_; ++b) { float v = Z[(size_t)b * J + j]; s += v; q += v * v; }
  float m  = s * (1.0f / B_);
  float vr = q * (1.0f / B_) - m * m;
  float rs = rsqrtf(vr + EPSF);
  for (int b = 0; b < B_; ++b) {
    float v = g[j] * (Z[(size_t)b * J + j] - m) * rs + be[j];
    Aout[(size_t)b * J + j] = fmaxf(v, 0.0f);
  }
}

// ---------------- host launch ----------------
extern "C" void kernel_launch(void* const* d_in, const int* in_sizes, int n_in,
                              void* d_out, int out_size, void* d_ws, size_t ws_size,
                              hipStream_t stream) {
  // inputs in setup_inputs() order (params dict flattened in insertion order)
  const float* obs   = (const float*)d_in[0];   // (B,S)
  const float* th    = (const float*)d_in[1];   // (B,VI)
  /* intensity d_in[2] unused by reference */
  const float* dsf   = (const float*)d_in[3];
  const float* w1    = (const float*)d_in[4];   // (512,208)
  const float* w2    = (const float*)d_in[6];   // (1024,512)
  const float* w3    = (const float*)d_in[8];   // (2048,1024)
  const float* fw1   = (const float*)d_in[10];  // (1024,2048)
  const float* fw2   = (const float*)d_in[12];  // (512,1024)
  const float* g_in  = (const float*)d_in[14];
  const float* be_in = (const float*)d_in[15];
  const float* g1    = (const float*)d_in[16];
  const float* be1   = (const float*)d_in[17];
  const float* g2    = (const float*)d_in[18];
  const float* be2   = (const float*)d_in[19];
  const float* g3    = (const float*)d_in[20];
  const float* be3   = (const float*)d_in[21];
  const float* g4    = (const float*)d_in[22];
  const float* be4   = (const float*)d_in[23];
  const float* g5    = (const float*)d_in[24];
  const float* be5   = (const float*)d_in[25];
  // (biases b1/b2/b3/fb1/fb2 cancel exactly under the following BatchNorms — skipped)

  char* ws = (char*)d_ws;
  size_t off = 0;
  auto alloc = [&](size_t bytes) { size_t o = off; off += (bytes + 255) & ~(size_t)255; return o; };
  unsigned short* W1b = (unsigned short*)(ws + alloc((size_t)O1_ * VIP_ * 2));
  unsigned short* W2b = (unsigned short*)(ws + alloc((size_t)O2_ * O1_ * 2));
  unsigned short* W3b = (unsigned short*)(ws + alloc((size_t)O3_ * O2_ * 2));
  float* meanB = (float*)(ws + alloc((size_t)O3_ * 4));
  float* rstdB = (float*)(ws + alloc((size_t)O3_ * 4));
  float* psumA = (float*)(ws + alloc((size_t)NSEG_ * O2_ * 4));
  float* psqA  = (float*)(ws + alloc((size_t)NSEG_ * O2_ * 4));
  float* psum3 = (float*)(ws + alloc((size_t)(P_ / 32) * O3_ * 4));   // 1024 segs
  float* psq3  = (float*)(ws + alloc((size_t)(P_ / 32) * O3_ * 4));
  unsigned* umax = (unsigned*)(ws + alloc((size_t)B_ * O3_ * 4));
  unsigned* umin = (unsigned*)(ws + alloc((size_t)B_ * O3_ * 4));
  float* Mbuf = (float*)(ws + alloc((size_t)B_ * O3_ * 4));
  float* Z1   = (float*)(ws + alloc((size_t)B_ * FJ1_ * 4));
  float* A1   = (float*)(ws + alloc((size_t)B_ * FJ1_ * 4));
  float* Z2   = (float*)(ws + alloc((size_t)B_ * FJ2_ * 4));
  // big arenas with lifetime aliasing
  char* arena1 = ws + alloc((size_t)P_ * O2_ * 4);  // 128MB: feat | Y1 | Y2
  char* arena2 = ws + alloc((size_t)P_ * O2_ * 2);  // 64MB:  X0 | X1 | X2
  float* feat = (float*)arena1;
  float* Y1   = (float*)arena1;
  float* Y2   = (float*)arena1;
  unsigned short* X0 = (unsigned short*)arena2;
  unsigned short* X1 = (unsigned short*)arena2;
  unsigned short* X2 = (unsigned short*)arena2;
  (void)ws_size; (void)in_sizes; (void)n_in; (void)out_size;

  const int T = 256;
  const float invP = 1.0f / (float)P_;
  const int chunk = P_ / NSEG_;   // 512

  // weights -> bf16 (w1 K-padded 208->224)
  wcvt_kernel<<<(O1_ * VIP_ + T - 1) / T, T, 0, stream>>>(w1, W1b, O1_, VI_, VIP_);
  wcvt_kernel<<<(O2_ * O1_ + T - 1) / T, T, 0, stream>>>(w2, W2b, O2_, O1_, O1_);
  wcvt_kernel<<<(O3_ * O2_ + T - 1) / T, T, 0, stream>>>(w3, W3b, O3_, O2_, O2_);

  // features + input BN -> X0 bf16 [p][224]
  feat_kernel<<<(P_ * VI_ + T - 1) / T, T, 0, stream>>>(obs, th, dsf, feat);
  stats_partial_kernel<<<dim3((VI_ + T - 1) / T, NSEG_), T, 0, stream>>>(feat, psumA, psqA, VI_, chunk);
  stats_final_kernel<<<(VI_ + T - 1) / T, T, 0, stream>>>(psumA, psqA, NSEG_, VI_, invP, meanB, rstdB);
  bnact_kernel<<<(int)(((size_t)P_ * VIP_ + T - 1) / T), T, 0, stream>>>(
      feat, meanB, rstdB, g_in, be_in, X0, VI_, VIP_, 0);

  // L1: (P x 224) x (512 x 224)^T -> Y1, BN+relu -> X1
  gemm_bf16_kernel<0><<<dim3(P_ / 64, O1_ / 256), T, 0, stream>>>(
      X0, W1b, Y1, nullptr, nullptr, nullptr, nullptr, VIP_, O1_);
  stats_partial_kernel<<<dim3(O1_ / T, NSEG_), T, 0, stream>>>(Y1, psumA, psqA, O1_, chunk);
  stats_final_kernel<<<O1_ / T, T, 0, stream>>>(psumA, psqA, NSEG_, O1_, invP, meanB, rstdB);
  bnact_kernel<<<(int)(((size_t)P_ * O1_) / T), T, 0, stream>>>(
      Y1, meanB, rstdB, g1, be1, X1, O1_, O1_, 1);

  // L2: -> Y2, BN+relu -> X2
  gemm_bf16_kernel<0><<<dim3(P_ / 64, O2_ / 256), T, 0, stream>>>(
      X1, W2b, Y2, nullptr, nullptr, nullptr, nullptr, O1_, O2_);
  stats_partial_kernel<<<dim3(O2_ / T, NSEG_), T, 0, stream>>>(Y2, psumA, psqA, O2_, chunk);
  stats_final_kernel<<<O2_ / T, T, 0, stream>>>(psumA, psqA, NSEG_, O2_, invP, meanB, rstdB);
  bnact_kernel<<<(int)(((size_t)P_ * O2_) / T), T, 0, stream>>>(
      Y2, meanB, rstdB, g2, be2, X2, O2_, O2_, 1);

  // L3: fused reduce epilogue (no 256MB materialization)
  init_minmax_kernel<<<(B_ * O3_ + T - 1) / T, T, 0, stream>>>(umax, umin, B_ * O3_);
  gemm_bf16_kernel<1><<<dim3(P_ / 64, O3_ / 256), T, 0, stream>>>(
      X2, W3b, nullptr, psum3, psq3, umax, umin, O2_, O3_);
  stats_final_kernel<<<O3_ / T, T, 0, stream>>>(psum3, psq3, P_ / 32, O3_, invP, meanB, rstdB);
  bnmax_kernel<<<(B_ * O3_ + T - 1) / T, T, 0, stream>>>(
      umax, umin, meanB, rstdB, g3, be3, Mbuf, O3_);

  // FC head (tiny)
  fc_kernel<<<(B_ * FJ1_ + T - 1) / T, T, 0, stream>>>(Mbuf, fw1, Z1, FK1_, FJ1_);
  bnrelu_c_kernel<<<(FJ1_ + T - 1) / T, T, 0, stream>>>(Z1, g4, be4, A1, FJ1_);
  fc_kernel<<<(B_ * FJ2_ + T - 1) / T, T, 0, stream>>>(A1, fw2, Z2, FK2_, FJ2_);
  bnrelu_c_kernel<<<(FJ2_ + T - 1) / T, T, 0, stream>>>(Z2, g5, be5, (float*)d_out, FJ2_);
}
